// IndraQuantumLayer_55886114455740
// MI455X (gfx1250) — compile-verified, matrix operand order checked by reading, not run on hardware
//
#include <hip/hip_runtime.h>
#include <hip/hip_bf16.h>

typedef __bf16 bf16_t;
typedef __attribute__((ext_vector_type(16))) __bf16 v16bf;
typedef __attribute__((ext_vector_type(8)))  float  v8f;
typedef unsigned int u32x4 __attribute__((ext_vector_type(4)));
typedef int          i32x4 __attribute__((ext_vector_type(4)));
typedef int          i32x8 __attribute__((ext_vector_type(8)));

#if __has_builtin(__builtin_amdgcn_tensor_load_to_lds) && \
    __has_builtin(__builtin_amdgcn_s_wait_tensorcnt)
#define USE_TDM 1
#else
#define USE_TDM 0
#endif

// ---------------------------------------------------------------------------
// helpers
// ---------------------------------------------------------------------------
__device__ __forceinline__ bf16_t f2b(float f) {
    unsigned u = __builtin_bit_cast(unsigned, f);
    unsigned r = u + 0x7fffu + ((u >> 16) & 1u);          // round-to-nearest-even
    unsigned short h = (unsigned short)(r >> 16);
    return __builtin_bit_cast(bf16_t, h);
}

union FragU { v16bf v; float4 q[2]; };

// Load a 16x32 (rows x K) bf16 fragment from a row-major [rows x ld] matrix in
// the CDNA5 16-bit A/B VGPR layout: lane r (0-15) holds row r; lanes 16-31 use
// K-octet offset +8.  Two contiguous 16-byte loads per lane.
__device__ __forceinline__ v16bf load_frag(const bf16_t* base, int ld, int lane) {
    FragU f;
    const int r  = lane & 15;
    const int ko = (lane & 16) ? 8 : 0;
    const bf16_t* p = base + (size_t)r * (size_t)ld + ko;
    f.q[0] = *(const float4*)(p);
    f.q[1] = *(const float4*)(p + 16);
    return f.v;
}

__device__ __forceinline__ v8f wmma_bf16(v16bf a, v16bf b, v8f c) {
    return __builtin_amdgcn_wmma_f32_16x16x32_bf16(false, a, false, b, (short)0, c,
                                                   false, false);
}

#if USE_TDM
// Issue one TDM DMA: 64 rows x 32 cols bf16 slab, row-major (stride K elems)
// from global into LDS at byte offset ldsOff (stored densely, 64B per row).
// D# layout per CDNA5 ISA ch.8 (2-D tile: groups 2/3 zero).
// Toolchain uses the 6-arg builtin form:
//   (u32x4 g0, i32x8 g1, i32x4, i32x4, i32x8, i32 cpol)
__device__ __forceinline__ void tdm_load_slab(const bf16_t* gsrc, unsigned ldsOff,
                                              int dim0, int dim1, int strideElems) {
    const unsigned long long ga = (unsigned long long)(size_t)gsrc;
    u32x4 g0;
    g0[0] = 1u;                                           // count=1 valid descriptor
    g0[1] = ldsOff;                                       // lds_addr (bytes)
    g0[2] = (unsigned)(ga & 0xffffffffu);                 // global_addr[31:0]
    g0[3] = (unsigned)((ga >> 32) & 0x01ffffffu) | (2u << 30);  // [56:32] | type=2
    const unsigned long long st0 = (unsigned long long)(unsigned)strideElems;
    i32x8 g1;
    g1[0] = (int)(1u << 16);                              // data_size=1 (2 bytes)
    g1[1] = (int)(((unsigned)dim0 & 0xffffu) << 16);      // tensor_dim0[15:0]
    g1[2] = (int)((((unsigned)dim0 >> 16) & 0xffffu) |
                  (((unsigned)dim1 & 0xffffu) << 16));    // dim0 hi | tensor_dim1 lo
    g1[3] = (int)(((((unsigned)dim1) >> 16) & 0xffffu) |
                  (32u << 16));                           // dim1 hi | tile_dim0=32
    g1[4] = (int)64u;                                     // tile_dim1=64, tile_dim2=0
    g1[5] = (int)(unsigned)(st0 & 0xffffffffu);           // tensor_dim0_stride lo
    g1[6] = (int)(unsigned)((st0 >> 32) & 0xffffu);       // stride hi | dim1_stride lo
    g1[7] = 0;
    const i32x4 z4 = {0, 0, 0, 0};
    const i32x8 z8 = {0, 0, 0, 0, 0, 0, 0, 0};
    __builtin_amdgcn_tensor_load_to_lds(g0, g1, z4, z4, z8, 0);
}
#endif

// ---------------------------------------------------------------------------
// fp32 -> bf16 convert
// ---------------------------------------------------------------------------
__global__ __launch_bounds__(256)
void convert_f32_bf16(const float* __restrict__ in, bf16_t* __restrict__ out,
                      long long n) {
    long long i      = (long long)blockIdx.x * blockDim.x + threadIdx.x;
    long long stride = (long long)gridDim.x * blockDim.x;
    for (; i < n; i += stride) out[i] = f2b(in[i]);
}

// ---------------------------------------------------------------------------
// WMMA GEMM: C[M,N] = A[M,K] * B[N,K]^T  (both row-major, contiguous K)
// A-slab (64x32) staged into LDS once per block per K-step via TDM, double
// buffered so the DMA of slab t+1 overlaps the WMMAs of slab t.  B fragments
// are per-wave unique and stream from global.  Each wave: 64 rows x 16 cols.
// Block = 8 waves = 128 columns.  Batched via blockIdx.z strides.
// ---------------------------------------------------------------------------
enum { EPI_SCALE_F32 = 0, EPI_BIAS_BF16 = 1, EPI_BIAS_RES_F32 = 2,
       EPI_BIAS_GELU_BF16 = 3 };

template <int EPI>
__global__ __launch_bounds__(256)
void gemm_bf16_wmma(const bf16_t* A, const bf16_t* Bm,
                    float* __restrict__ outF, bf16_t* __restrict__ outB,
                    const float* __restrict__ bias,
                    const float* __restrict__ resid,
                    int M, int N, int K, float scale,
                    long long strideA, long long strideB, long long strideC) {
    __shared__ bf16_t tileA[2][64][32];                  // 8 KB double buffer

    const int lane = threadIdx.x & 31;
    const int wave = threadIdx.x >> 5;
    const int n0   = (blockIdx.x * 8 + wave) * 16;
    const int m0   = blockIdx.y * 64;
    const int z    = blockIdx.z;

    const bf16_t* aRow = A  + (size_t)z * (size_t)strideA + (size_t)m0 * (size_t)K;
    const bf16_t* bRow = Bm + (size_t)z * (size_t)strideB + (size_t)n0 * (size_t)K;

    v8f acc[4];
#pragma unroll
    for (int i = 0; i < 4; ++i)
#pragma unroll
        for (int e = 0; e < 8; ++e) acc[i][e] = 0.0f;

#if USE_TDM
    const unsigned ldsBase = (unsigned)(size_t)&tileA[0][0][0];
    if (wave == 0)
        tdm_load_slab(aRow, ldsBase, K, M - m0, K);      // slab 0 -> buf 0
#endif

    int buf = 0;
    for (int k0 = 0; k0 < K; k0 += 32, buf ^= 1) {
#if USE_TDM
        if (wave == 0) __builtin_amdgcn_s_wait_tensorcnt(0);  // slab k0 landed
        __syncthreads();            // slab visible; prev reads of other buf done
        if (wave == 0 && k0 + 32 < K)
            tdm_load_slab(aRow + k0 + 32, ldsBase + (unsigned)(buf ^ 1) * 4096u,
                          K - (k0 + 32), M - m0, K);     // overlap next DMA
#endif
        v16bf bf = load_frag(bRow + k0, K, lane);
        __builtin_prefetch(bRow + k0 + 64, 0, 1);        // global_prefetch_b8
#pragma unroll
        for (int i = 0; i < 4; ++i) {
            v16bf af;
#if USE_TDM
            {   // A fragment from LDS slab (ds_load_b128 x2)
                FragU f;
                const bf16_t* p =
                    &tileA[buf][i * 16 + (lane & 15)][(lane & 16) ? 8 : 0];
                f.q[0] = *(const float4*)(p);
                f.q[1] = *(const float4*)(p + 16);
                af = f.v;
            }
#else
            af = load_frag(aRow + (size_t)(i * 16) * (size_t)K + k0, K, lane);
#endif
            acc[i] = wmma_bf16(af, bf, acc[i]);
        }
    }

    const int col   = n0 + (lane & 15);
    const int rbase = m0 + ((lane & 16) ? 8 : 0);
    float bv = 0.0f;
    if constexpr (EPI != EPI_SCALE_F32) bv = bias[col];

#pragma unroll
    for (int i = 0; i < 4; ++i) {
#pragma unroll
        for (int e = 0; e < 8; ++e) {
            const int row = rbase + i * 16 + e;
            const size_t idx = (size_t)z * (size_t)strideC +
                               (size_t)row * (size_t)N + (size_t)col;
            const float v = acc[i][e];
            if constexpr (EPI == EPI_SCALE_F32) {
                outF[idx] = v * scale;
            } else if constexpr (EPI == EPI_BIAS_BF16) {
                outB[idx] = f2b(v + bv);
            } else if constexpr (EPI == EPI_BIAS_RES_F32) {
                outF[idx] = v + bv + resid[idx];
            } else {  // EPI_BIAS_GELU_BF16, exact GELU
                const float t = v + bv;
                outB[idx] = f2b(0.5f * t * (1.0f + erff(t * 0.70710678118f)));
            }
        }
    }
}

// ---------------------------------------------------------------------------
// qkv [B*S, 3D] -> q[B,H,S,dh], k[B,H,S,dh], vT[B,H,dh,S]   (dh = 64)
// ---------------------------------------------------------------------------
__global__ __launch_bounds__(256)
void split_qkv_kernel(const bf16_t* __restrict__ qkv, bf16_t* __restrict__ q,
                      bf16_t* __restrict__ k, bf16_t* __restrict__ vT,
                      int B, int S, int H) {
    const int dh = 64;
    const int D  = H * dh;
    const long long n = (long long)B * S * D;
    long long i      = (long long)blockIdx.x * blockDim.x + threadIdx.x;
    long long stride = (long long)gridDim.x * blockDim.x;
    for (; i < n; i += stride) {
        const int d  = (int)(i % dh);
        long long t  = i / dh;
        const int hh = (int)(t % H);
        t /= H;
        const int s = (int)(t % S);
        const int b = (int)(t / S);
        const size_t src = ((size_t)b * S + s) * (size_t)(3 * D) + hh * dh + d;
        const size_t qi  = (((size_t)b * H + hh) * S + s) * dh + d;
        const size_t vti = (((size_t)b * H + hh) * dh + d) * (size_t)S + s;
        q[qi]  = qkv[src];
        k[qi]  = qkv[src + D];
        vT[vti] = qkv[src + 2 * D];
    }
}

// ---------------------------------------------------------------------------
// Flash attention with hopping bias + causal mask.
// grid (S/64, H, B), 4 waves/block, 16 q-rows per wave, dh = 64.
// ---------------------------------------------------------------------------
__global__ __launch_bounds__(128)
void flash_attn_kernel(const bf16_t* __restrict__ q, const bf16_t* __restrict__ k,
                       const bf16_t* __restrict__ vT, const float* __restrict__ hop,
                       bf16_t* __restrict__ out, int S, int H) {
    const int lane = threadIdx.x & 31;
    const int wave = threadIdx.x >> 5;
    const int h  = blockIdx.y;
    const int b  = blockIdx.z;
    const int q0 = blockIdx.x * 64 + wave * 16;
    const int D  = H * 64;

    const bf16_t* qh = q  + ((size_t)(b * H + h) * (size_t)S) * 64;
    const bf16_t* kh = k  + ((size_t)(b * H + h) * (size_t)S) * 64;
    const bf16_t* vh = vT + ((size_t)(b * H + h) * 64) * (size_t)S;
    const float*  hb = hop + (size_t)b * (size_t)S * (size_t)S;

    __shared__ bf16_t plds[4][16][32];   // per-wave probability tile

    const v16bf qf0 = load_frag(qh + (size_t)q0 * 64,      64, lane);
    const v16bf qf1 = load_frag(qh + (size_t)q0 * 64 + 32, 64, lane);

    v8f o[4];
    float m[8], l[8];
#pragma unroll
    for (int i = 0; i < 4; ++i)
#pragma unroll
        for (int e = 0; e < 8; ++e) o[i][e] = 0.0f;
#pragma unroll
    for (int e = 0; e < 8; ++e) { m[e] = -3.0e38f; l[e] = 0.0f; }

    const int col16 = lane & 15;
    const int rOff  = (lane & 16) ? 8 : 0;

    for (int kb = 0; kb < q0 + 16; kb += 32) {
        v8f s0, s1;
#pragma unroll
        for (int e = 0; e < 8; ++e) { s0[e] = 0.0f; s1[e] = 0.0f; }
        {
            v16bf kf0 = load_frag(kh + (size_t)kb * 64,      64, lane);
            v16bf kf1 = load_frag(kh + (size_t)kb * 64 + 32, 64, lane);
            s0 = wmma_bf16(qf0, kf0, s0);
            s0 = wmma_bf16(qf1, kf1, s0);
            kf0 = load_frag(kh + (size_t)(kb + 16) * 64,      64, lane);
            kf1 = load_frag(kh + (size_t)(kb + 16) * 64 + 32, 64, lane);
            s1 = wmma_bf16(qf0, kf0, s1);
            s1 = wmma_bf16(qf1, kf1, s1);
        }

        float p0[8], p1[8], mx[8];
#pragma unroll
        for (int e = 0; e < 8; ++e) {
            const int row = q0 + rOff + e;
            const int c0  = kb + col16;
            const int c1  = kb + 16 + col16;
            float a0 = s0[e] * 0.125f + hb[(size_t)row * S + c0];
            float a1 = s1[e] * 0.125f + hb[(size_t)row * S + c1];
            if (c0 > row) a0 = -3.0e38f;
            if (c1 > row) a1 = -3.0e38f;
            p0[e] = a0;
            p1[e] = a1;
            mx[e] = fmaxf(a0, a1);
        }
        // row max over the 16-lane half (rows live one-per-VGPR across lanes)
#pragma unroll
        for (int msk = 1; msk < 16; msk <<= 1)
#pragma unroll
            for (int e = 0; e < 8; ++e)
                mx[e] = fmaxf(mx[e], __shfl_xor(mx[e], msk, 32));

        float alpha[8], rs[8];
#pragma unroll
        for (int e = 0; e < 8; ++e) {
            const float mn = fmaxf(m[e], mx[e]);
            alpha[e] = __expf(m[e] - mn);
            m[e]     = mn;
            p0[e] = __expf(p0[e] - mn);
            p1[e] = __expf(p1[e] - mn);
            rs[e] = p0[e] + p1[e];
        }
#pragma unroll
        for (int msk = 1; msk < 16; msk <<= 1)
#pragma unroll
            for (int e = 0; e < 8; ++e)
                rs[e] += __shfl_xor(rs[e], msk, 32);
#pragma unroll
        for (int e = 0; e < 8; ++e) l[e] = l[e] * alpha[e] + rs[e];
#pragma unroll
        for (int i = 0; i < 4; ++i)
#pragma unroll
            for (int e = 0; e < 8; ++e) o[i][e] *= alpha[e];

        // C-layout -> A-layout through a 1KB per-wave LDS tile
#pragma unroll
        for (int e = 0; e < 8; ++e) {
            plds[wave][rOff + e][col16]      = f2b(p0[e]);
            plds[wave][rOff + e][16 + col16] = f2b(p1[e]);
        }
        v16bf pf;
        {
            FragU f;
            const int ko = (lane & 16) ? 8 : 0;
            const bf16_t* pb = &plds[wave][lane & 15][ko];
            f.q[0] = *(const float4*)(pb);
            f.q[1] = *(const float4*)(pb + 16);
            pf = f.v;
        }

#pragma unroll
        for (int i = 0; i < 4; ++i) {
            v16bf vf = load_frag(vh + (size_t)(i * 16) * (size_t)S + kb, S, lane);
            o[i] = wmma_bf16(pf, vf, o[i]);
        }
    }

    float inv[8];
#pragma unroll
    for (int e = 0; e < 8; ++e) inv[e] = 1.0f / l[e];
#pragma unroll
    for (int i = 0; i < 4; ++i)
#pragma unroll
        for (int e = 0; e < 8; ++e) {
            const int row = q0 + rOff + e;
            out[((size_t)b * S + row) * (size_t)D + h * 64 + i * 16 + col16] =
                f2b(o[i][e] * inv[e]);
        }
}

// ---------------------------------------------------------------------------
// LayerNorm: one row (D=1024) per block; writes f32 (+ optional bf16 copy)
// ---------------------------------------------------------------------------
__global__ __launch_bounds__(256)
void layernorm_kernel(const float* __restrict__ x, const float* __restrict__ g,
                      const float* __restrict__ be, float* __restrict__ yF,
                      bf16_t* __restrict__ yB, int D) {
    const int row = blockIdx.x;
    const float* xr = x + (size_t)row * D;
    float s = 0.0f, s2 = 0.0f;
    for (int i = threadIdx.x; i < D; i += 256) {
        const float v = xr[i];
        s += v;
        s2 += v * v;
    }
    __shared__ float red0[256];
    __shared__ float red1[256];
    red0[threadIdx.x] = s;
    red1[threadIdx.x] = s2;
    __syncthreads();
    for (int st = 128; st > 0; st >>= 1) {
        if ((int)threadIdx.x < st) {
            red0[threadIdx.x] += red0[threadIdx.x + st];
            red1[threadIdx.x] += red1[threadIdx.x + st];
        }
        __syncthreads();
    }
    const float mu   = red0[0] / (float)D;
    const float var  = red1[0] / (float)D - mu * mu;
    const float rstd = rsqrtf(var + 1e-5f);
    for (int i = threadIdx.x; i < D; i += 256) {
        const float v = (xr[i] - mu) * rstd * g[i] + be[i];
        yF[(size_t)row * D + i] = v;
        if (yB) yB[(size_t)row * D + i] = f2b(v);
    }
}

// ---------------------------------------------------------------------------
// host
// ---------------------------------------------------------------------------
extern "C" void kernel_launch(void* const* d_in, const int* in_sizes, int n_in,
                              void* d_out, int out_size, void* d_ws, size_t ws_size,
                              hipStream_t stream) {
    (void)in_sizes; (void)n_in; (void)out_size; (void)ws_size;
    const float* x     = (const float*)d_in[0];
    const float* imag  = (const float*)d_in[1];
    const float* w_qkv = (const float*)d_in[2];
    const float* b_qkv = (const float*)d_in[3];
    const float* w_o   = (const float*)d_in[4];
    const float* b_o   = (const float*)d_in[5];
    const float* ln1g  = (const float*)d_in[6];
    const float* ln1b  = (const float*)d_in[7];
    const float* w1    = (const float*)d_in[8];
    const float* b1    = (const float*)d_in[9];
    const float* w2    = (const float*)d_in[10];
    const float* b2    = (const float*)d_in[11];
    const float* ln2g  = (const float*)d_in[12];
    const float* ln2b  = (const float*)d_in[13];

    const int B = 4, S = 2048, D = 1024, H = 16;
    const long long BS = (long long)B * S;

    // ---- workspace layout (lifetime-based reuse; peak ~216 MB) ----
    char* ws = (char*)d_ws;
    const size_t MB = 1ull << 20;
    float*  hopping = (float*)(ws + 0);            // 64 MB  (B*S*S f32)
    bf16_t* hbuf    = (bf16_t*)(ws + 0);           // reuse: FFN hidden (64 MB bf16)
    bf16_t* qkv     = (bf16_t*)(ws + 64 * MB);     // 48 MB
    float*  pre1    = (float*)(ws + 64 * MB);      // reuse: pre-LN1 (32 MB)
    float*  pre2    = (float*)(ws + 64 * MB);      // reuse: pre-LN2 (32 MB)
    bf16_t* qb      = (bf16_t*)(ws + 112 * MB);    // 16 MB
    bf16_t* kb      = (bf16_t*)(ws + 128 * MB);    // 16 MB
    bf16_t* vTb     = (bf16_t*)(ws + 144 * MB);    // 16 MB
    float*  x1f     = (float*)(ws + 112 * MB);     // reuse q+k (32 MB)
    bf16_t* x1b     = (bf16_t*)(ws + 144 * MB);    // reuse vT (16 MB)
    bf16_t* xbf     = (bf16_t*)(ws + 160 * MB);    // 16 MB
    bf16_t* attn    = (bf16_t*)(ws + 160 * MB);    // reuse xbf (16 MB)
    bf16_t* imgbf   = (bf16_t*)(ws + 176 * MB);    // 16 MB
    bf16_t* wqkvb   = (bf16_t*)(ws + 192 * MB);    // 6 MB
    bf16_t* wob     = (bf16_t*)(ws + 198 * MB);    // 2 MB
    bf16_t* w1b     = (bf16_t*)(ws + 200 * MB);    // 8 MB
    bf16_t* w2b     = (bf16_t*)(ws + 208 * MB);    // 8 MB  -> 216 MB total

    // ---- 1: converts ----
    convert_f32_bf16<<<2048, 256, 0, stream>>>(x,     xbf,   BS * D);
    convert_f32_bf16<<<2048, 256, 0, stream>>>(imag,  imgbf, BS * D);
    convert_f32_bf16<<<2048, 256, 0, stream>>>(w_qkv, wqkvb, (long long)3 * D * D);
    convert_f32_bf16<<<2048, 256, 0, stream>>>(w_o,   wob,   (long long)D * D);
    convert_f32_bf16<<<2048, 256, 0, stream>>>(w1,    w1b,   (long long)4 * D * D);
    convert_f32_bf16<<<2048, 256, 0, stream>>>(w2,    w2b,   (long long)4 * D * D);

    // ---- 2: hopping = imag @ imag^T / sqrt(D)  (batched S x S, K = D) ----
    {
        dim3 g(S / 128, S / 64, B);
        gemm_bf16_wmma<EPI_SCALE_F32><<<g, 256, 0, stream>>>(
            imgbf, imgbf, hopping, nullptr, nullptr, nullptr,
            S, S, D, 0.03125f,
            (long long)S * D, (long long)S * D, (long long)S * S);
    }

    // ---- 3: QKV projection ----
    {
        dim3 g((3 * D) / 128, (int)(BS / 64), 1);
        gemm_bf16_wmma<EPI_BIAS_BF16><<<g, 256, 0, stream>>>(
            xbf, wqkvb, nullptr, qkv, b_qkv, nullptr,
            (int)BS, 3 * D, D, 1.0f, 0, 0, 0);
    }

    // ---- 4: split / per-head reshape (v transposed) ----
    split_qkv_kernel<<<8192, 256, 0, stream>>>(qkv, qb, kb, vTb, B, S, H);

    // ---- 5: flash attention (+hopping bias, causal) ----
    {
        dim3 g(S / 64, H, B);
        flash_attn_kernel<<<g, 128, 0, stream>>>(qb, kb, vTb, hopping, attn, S, H);
    }

    // ---- 6: output projection + residual ----
    {
        dim3 g(D / 128, (int)(BS / 64), 1);
        gemm_bf16_wmma<EPI_BIAS_RES_F32><<<g, 256, 0, stream>>>(
            attn, wob, pre1, nullptr, b_o, x,
            (int)BS, D, D, 1.0f, 0, 0, 0);
    }

    // ---- 7: LN1 -> x1 (f32 + bf16) ----
    layernorm_kernel<<<(int)BS, 256, 0, stream>>>(pre1, ln1g, ln1b, x1f, x1b, D);

    // ---- 8: FFN1 (bias + exact GELU) ----
    {
        dim3 g((4 * D) / 128, (int)(BS / 64), 1);
        gemm_bf16_wmma<EPI_BIAS_GELU_BF16><<<g, 256, 0, stream>>>(
            x1b, w1b, nullptr, hbuf, b1, nullptr,
            (int)BS, 4 * D, D, 1.0f, 0, 0, 0);
    }

    // ---- 9: FFN2 + residual ----
    {
        dim3 g(D / 128, (int)(BS / 64), 1);
        gemm_bf16_wmma<EPI_BIAS_RES_F32><<<g, 256, 0, stream>>>(
            hbuf, w2b, pre2, nullptr, b2, x1f,
            (int)BS, D, 4 * D, 1.0f, 0, 0, 0);
    }

    // ---- 10: LN2 -> d_out ----
    layernorm_kernel<<<(int)BS, 256, 0, stream>>>(pre2, ln2g, ln2b,
                                                  (float*)d_out, nullptr, D);
}
